// BiLevelRoutingAttention_3951369912844
// MI455X (gfx1250) — matrix operand order repaired and checked by reading, not compiled, not run on hardware
//
#include <hip/hip_runtime.h>
#include <hip/hip_bf16.h>
#include <stdint.h>

typedef __attribute__((ext_vector_type(16))) __bf16 v16bf;
typedef __attribute__((ext_vector_type(8)))  float  v8f;

#define BATCH   4
#define HH      56
#define WWIDTH  56
#define CH      256
#define NH      8
#define HD      32
#define LWIN    3136            // 56*56 windowed tokens per batch
#define LSEQ    3137            // + cls token
#define LPAD    3168            // 99 * 32
#define KB32    (LPAD / 32)     // 99 key blocks of 32
#define ROWS    12548           // B*H*W + B cls rows
#define ROWSPAD 12576           // 393 * 32
#define WINNUM  49
#define QTILES  197             // ceil(3137/16)
#define SCALEQ  0.0625f         // 256^-0.5
#define NEGINF  (-1.0e30f)

// ---------------------------------------------------------------- WMMA helpers

__device__ __forceinline__ v8f zero8() {
  v8f z;
#pragma unroll
  for (int i = 0; i < 8; ++i) z[i] = 0.0f;
  return z;
}

__device__ __forceinline__ v8f wmma_bf16(v16bf a, v16bf b, v8f c) {
  // D(16x16 f32) = A(16x32 bf16) * B(32x16 bf16) + C
  return __builtin_amdgcn_wmma_f32_16x16x32_bf16(false, a, false, b,
                                                 (short)0, c, false, false);
}

// A fragment (16x32, row-major source, ld = elements per row).
// lanes 0-15: row = lane, K = {0..7, 16..23}; lanes 16-31: row = lane-16,
// K = {8..15, 24..31}  (CDNA5 ISA 7.12.2, 16-bit A 16x32)
__device__ __forceinline__ v16bf load_frag_a(const __bf16* base, int ld, int lane) {
  int m = lane & 15, h = lane >> 4;
  const __bf16* p0 = base + m * ld + h * 8;
  const __bf16* p1 = p0 + 16;
  v16bf r;
#pragma unroll
  for (int i = 0; i < 8; ++i) { r[i] = p0[i]; r[8 + i] = p1[i]; }
  return r;
}

// B fragment (32x16) from a *transposed* source Bt (N x K row-major):
// lane holds column n = lane&15, K = {h*8..h*8+7, 16+h*8..16+h*8+7}, h = lane>>4.
__device__ __forceinline__ v16bf load_frag_b(const __bf16* base, int ld, int lane) {
  int n = lane & 15, h = lane >> 4;
  const __bf16* p0 = base + n * ld + h * 8;
  const __bf16* p1 = p0 + 16;
  v16bf r;
#pragma unroll
  for (int i = 0; i < 8; ++i) { r[i] = p0[i]; r[8 + i] = p1[i]; }
  return r;
}

// ---------------------------------------------------------------- prep kernels

__global__ void prep_a_kernel(const float* __restrict__ x,
                              const float* __restrict__ cls,
                              __bf16* __restrict__ A) {
  int idx = blockIdx.x * 256 + threadIdx.x;
  if (idx >= ROWSPAD * CH) return;
  int r = idx >> 8, c = idx & 255;
  float v = 0.0f;
  if (r < BATCH * LWIN)      v = x[idx];
  else if (r < ROWS)         v = cls[(r - BATCH * LWIN) * CH + c];
  A[idx] = (__bf16)v;
}

__global__ void prep_w_kernel(const float* __restrict__ Wqkv,
                              const float* __restrict__ Wo,
                              __bf16* __restrict__ WqkvT,
                              __bf16* __restrict__ WoT) {
  int idx = blockIdx.x * 256 + threadIdx.x;
  const int NQKV = 768 * 256;
  if (idx < NQKV) {
    int n = idx >> 8, k = idx & 255;
    WqkvT[idx] = (__bf16)Wqkv[k * 768 + n];
  } else {
    int j = idx - NQKV;
    if (j < 256 * 256) {
      int n = j >> 8, k = j & 255;
      WoT[j] = (__bf16)Wo[k * 256 + n];
    }
  }
}

// ---------------------------------------------------------------- QKV GEMM
// 32x64 tile per wave, double-buffered k-loop (loads of step k+1 overlap
// the 8 WMMAs of step k).
__global__ void __launch_bounds__(128)
gemm_qkv_kernel(const __bf16* __restrict__ A, const __bf16* __restrict__ Bt,
                const float* __restrict__ bias,
                __bf16* __restrict__ Qb, __bf16* __restrict__ Kb,
                __bf16* __restrict__ Vt, float* __restrict__ Vsp) {
  int lane = threadIdx.x & 31;
  int wv   = threadIdx.x >> 5;
  int row0 = blockIdx.x * 32;
  int col0 = (blockIdx.y * 4 + wv) * 64;

  v8f acc[2][4];
#pragma unroll
  for (int m = 0; m < 2; ++m)
#pragma unroll
    for (int j = 0; j < 4; ++j) acc[m][j] = zero8();

  const __bf16* A0 = A + row0 * CH;
  const __bf16* A1 = A + (row0 + 16) * CH;

  v16bf a0 = load_frag_a(A0, CH, lane);
  v16bf a1 = load_frag_a(A1, CH, lane);
  v16bf b0 = load_frag_b(Bt + (col0 +  0) * CH, CH, lane);
  v16bf b1 = load_frag_b(Bt + (col0 + 16) * CH, CH, lane);
  v16bf b2 = load_frag_b(Bt + (col0 + 32) * CH, CH, lane);
  v16bf b3 = load_frag_b(Bt + (col0 + 48) * CH, CH, lane);

  for (int kk = 0; kk < CH; kk += 32) {
    v16bf a0n = a0, a1n = a1, b0n = b0, b1n = b1, b2n = b2, b3n = b3;
    int kn = kk + 32;
    if (kn < CH) {                       // issue next-step loads first
      a0n = load_frag_a(A0 + kn, CH, lane);
      a1n = load_frag_a(A1 + kn, CH, lane);
      b0n = load_frag_b(Bt + (col0 +  0) * CH + kn, CH, lane);
      b1n = load_frag_b(Bt + (col0 + 16) * CH + kn, CH, lane);
      b2n = load_frag_b(Bt + (col0 + 32) * CH + kn, CH, lane);
      b3n = load_frag_b(Bt + (col0 + 48) * CH + kn, CH, lane);
    }
    acc[0][0] = wmma_bf16(a0, b0, acc[0][0]);
    acc[1][0] = wmma_bf16(a1, b0, acc[1][0]);
    acc[0][1] = wmma_bf16(a0, b1, acc[0][1]);
    acc[1][1] = wmma_bf16(a1, b1, acc[1][1]);
    acc[0][2] = wmma_bf16(a0, b2, acc[0][2]);
    acc[1][2] = wmma_bf16(a1, b2, acc[1][2]);
    acc[0][3] = wmma_bf16(a0, b3, acc[0][3]);
    acc[1][3] = wmma_bf16(a1, b3, acc[1][3]);
    a0 = a0n; a1 = a1n; b0 = b0n; b1 = b1n; b2 = b2n; b3 = b3n;
  }

  int half = lane >> 4, nloc = lane & 15;
#pragma unroll
  for (int j = 0; j < 4; ++j) {
    int n = col0 + j * 16 + nloc;
    float bv = bias[n];
    int which = n >> 8;                 // 0=q 1=k 2=v (uniform per wave tile)
    int cc = n & 255;
    int head = cc >> 5, hd = cc & 31;
#pragma unroll
    for (int mi = 0; mi < 2; ++mi) {
#pragma unroll
      for (int i = 0; i < 8; ++i) {
        int r = row0 + mi * 16 + i + 8 * half;
        if (r >= ROWS) continue;
        float val = acc[mi][j][i] + bv;
        int b, t;
        if (r < BATCH * LWIN) {
          b = r / LWIN;
          int pix = r - b * LWIN;
          int y = pix / WWIDTH, xx = pix - y * WWIDTH;
          t = (((y >> 3) * 7 + (xx >> 3)) << 6) + ((y & 7) << 3) + (xx & 7);
        } else { b = r - BATCH * LWIN; t = LWIN; }
        int bh = b * NH + head;
        if (which == 0)      Qb[(bh * LPAD + t) * HD + hd] = (__bf16)(val * SCALEQ);
        else if (which == 1) Kb[(bh * LPAD + t) * HD + hd] = (__bf16)val;
        else {
          Vt[(bh * HD + hd) * LPAD + t] = (__bf16)val;
          if (r < BATCH * LWIN) Vsp[r * CH + cc] = val;
        }
      }
    }
  }
}

// ---------------------------------------------------------------- lepe conv
__global__ void lepe_kernel(const float* __restrict__ Vsp,
                            const float* __restrict__ lk,
                            const float* __restrict__ lb,
                            float* __restrict__ lepe) {
  int idx = blockIdx.x * 256 + threadIdx.x;      // BATCH*LWIN*CH threads exactly
  int c = idx & 255, p = idx >> 8;
  int b = p / LWIN, pix = p - b * LWIN;
  int y = pix / WWIDTH, x = pix - y * WWIDTH;
  float acc = lb[c];
#pragma unroll
  for (int dy = -1; dy <= 1; ++dy)
#pragma unroll
    for (int dx = -1; dx <= 1; ++dx) {
      int yy = y + dy, xx = x + dx;
      if (yy >= 0 && yy < HH && xx >= 0 && xx < WWIDTH)
        acc += Vsp[((b * LWIN) + yy * WWIDTH + xx) * CH + c] *
               lk[((dy + 1) * 3 + (dx + 1)) * CH + c];
    }
  lepe[idx] = acc;
}

// ---------------------------------------------------------------- attention
__global__ void __launch_bounds__(256)
attn_kernel(const __bf16* __restrict__ Qb, const __bf16* __restrict__ Kb,
            const __bf16* __restrict__ Vt, const unsigned char* __restrict__ mask,
            float* __restrict__ AO) {
  __shared__ __bf16 pbuf[8][16][32];             // per-wave P transpose buffer
  int lane = threadIdx.x & 31;
  int wv   = threadIdx.x >> 5;
  int gw   = blockIdx.x * 8 + wv;
  if (gw >= BATCH * NH * QTILES) return;
  int qt = gw % QTILES;
  int bh = gw / QTILES;
  int b  = bh / NH;
  int half = lane >> 4, nloc = lane & 15;

  int  qb   = (qt < 196) ? (qt >> 2) : WINNUM;   // query window block (uniform)
  bool qcls = (qb == WINNUM);
  const unsigned char* mrow = mask + (b * WINNUM + (qcls ? 0 : qb)) * WINNUM;

  v16bf qf = load_frag_a(Qb + (bh * LPAD + qt * 16) * HD, HD, lane);

  v8f o0 = zero8(), o1 = zero8();
  float mrun[8], lrun[8];
#pragma unroll
  for (int i = 0; i < 8; ++i) { mrun[i] = NEGINF; lrun[i] = 0.0f; }

  for (int kb = 0; kb < KB32; ++kb) {
    int k0 = kb * 32, k1 = k0 + 16;
    bool live0 = (k0 < LWIN) ? (qcls || mrow[k0 >> 6]) : (k0 == LWIN);
    bool live1 = (k1 < LWIN) ? (qcls || mrow[k1 >> 6]) : (k1 == LWIN);
    if (!live0 && !live1) continue;              // skip dead block entirely

    // Issue all global loads for this block up front: V frags are consumed
    // last (after softmax), so their latency hides under the exp/shuffle work.
    const __bf16* vbase = Vt + (bh * HD) * LPAD + k0;
    v16bf vf0 = load_frag_b(vbase, LPAD, lane);           // hd 0..15
    v16bf vf1 = load_frag_b(vbase + 16 * LPAD, LPAD, lane); // hd 16..31
    v16bf kf0 = qf, kf1 = qf;
    if (live0) kf0 = load_frag_b(Kb + (bh * LPAD + k0) * HD, HD, lane);
    if (live1) kf1 = load_frag_b(Kb + (bh * LPAD + k1) * HD, HD, lane);

    float s0[8], s1[8];
    if (live0) {
      v8f s = wmma_bf16(qf, kf0, zero8());
      bool cut = (k0 == LWIN) && (nloc != 0);    // cls tile: only key 3136 valid
#pragma unroll
      for (int i = 0; i < 8; ++i) s0[i] = cut ? NEGINF : s[i];
    } else {
#pragma unroll
      for (int i = 0; i < 8; ++i) s0[i] = NEGINF;
    }
    if (live1) {
      v8f s = wmma_bf16(qf, kf1, zero8());
      bool cut = (k1 == LWIN) && (nloc != 0);
#pragma unroll
      for (int i = 0; i < 8; ++i) s1[i] = cut ? NEGINF : s[i];
    } else {
#pragma unroll
      for (int i = 0; i < 8; ++i) s1[i] = NEGINF;
    }

    if (kb + 1 < KB32)                           // hint next K block toward L0
      __builtin_prefetch(Kb + (bh * LPAD + (kb + 1) * 32) * HD, 0, 1);

    // online softmax: row stats live redundantly in all 16 lanes of a half
#pragma unroll
    for (int i = 0; i < 8; ++i) {
      float mloc = fmaxf(s0[i], s1[i]);
#pragma unroll
      for (int d = 1; d < 16; d <<= 1) mloc = fmaxf(mloc, __shfl_xor(mloc, d, 32));
      float mnew  = fmaxf(mrun[i], mloc);
      float alpha = __expf(mrun[i] - mnew);
      float p0 = __expf(s0[i] - mnew);
      float p1 = __expf(s1[i] - mnew);
      float ps = p0 + p1;
#pragma unroll
      for (int d = 1; d < 16; d <<= 1) ps += __shfl_xor(ps, d, 32);
      lrun[i] = lrun[i] * alpha + ps;
      mrun[i] = mnew;
      o0[i] *= alpha;
      o1[i] *= alpha;
      pbuf[wv][i + 8 * half][nloc]      = (__bf16)p0;   // transpose via LDS
      pbuf[wv][i + 8 * half][16 + nloc] = (__bf16)p1;
    }
    asm volatile("s_wait_dscnt 0" ::: "memory");        // P stores -> P frag loads

    v16bf pf = load_frag_a(&pbuf[wv][0][0], 32, lane);  // P as 16x32 A-matrix
    o0 = wmma_bf16(pf, vf0, o0);
    o1 = wmma_bf16(pf, vf1, o1);
  }

#pragma unroll
  for (int i = 0; i < 8; ++i) {
    float inv = 1.0f / (lrun[i] + 1e-20f);
    int q = qt * 16 + i + 8 * half;
    if (q < LSEQ) {
      AO[(bh * LPAD + q) * HD + nloc]      = o0[i] * inv;
      AO[(bh * LPAD + q) * HD + 16 + nloc] = o1[i] * inv;
    }
  }
}

// ---------------------------------------------------------------- gather+lepe
__global__ void build_a2_kernel(const float* __restrict__ AO,
                                const float* __restrict__ lepe,
                                __bf16* __restrict__ A2) {
  int idx = blockIdx.x * 256 + threadIdx.x;
  if (idx >= ROWSPAD * CH) return;
  int r = idx >> 8, c = idx & 255;
  int head = c >> 5, hd = c & 31;
  float val = 0.0f;
  if (r < BATCH * LWIN) {
    int b = r / LWIN;
    int pix = r - b * LWIN;
    int y = pix / WWIDTH, x = pix - y * WWIDTH;
    int t = (((y >> 3) * 7 + (x >> 3)) << 6) + ((y & 7) << 3) + (x & 7);
    val = AO[((b * NH + head) * LPAD + t) * HD + hd] + lepe[idx];
  } else if (r < ROWS) {
    int b = r - BATCH * LWIN;
    val = AO[((b * NH + head) * LPAD + LWIN) * HD + hd];
  }
  A2[idx] = (__bf16)val;
}

// ---------------------------------------------------------------- output GEMM
__global__ void __launch_bounds__(128)
gemm_out_kernel(const __bf16* __restrict__ A, const __bf16* __restrict__ Bt,
                const float* __restrict__ bias, float* __restrict__ out) {
  int lane = threadIdx.x & 31;
  int wv   = threadIdx.x >> 5;
  int row0 = blockIdx.x * 32;
  int col0 = wv * 64;

  v8f acc[2][4];
#pragma unroll
  for (int m = 0; m < 2; ++m)
#pragma unroll
    for (int j = 0; j < 4; ++j) acc[m][j] = zero8();

  const __bf16* A0 = A + row0 * CH;
  const __bf16* A1 = A + (row0 + 16) * CH;

  v16bf a0 = load_frag_a(A0, CH, lane);
  v16bf a1 = load_frag_a(A1, CH, lane);
  v16bf b0 = load_frag_b(Bt + (col0 +  0) * CH, CH, lane);
  v16bf b1 = load_frag_b(Bt + (col0 + 16) * CH, CH, lane);
  v16bf b2 = load_frag_b(Bt + (col0 + 32) * CH, CH, lane);
  v16bf b3 = load_frag_b(Bt + (col0 + 48) * CH, CH, lane);

  for (int kk = 0; kk < CH; kk += 32) {
    v16bf a0n = a0, a1n = a1, b0n = b0, b1n = b1, b2n = b2, b3n = b3;
    int kn = kk + 32;
    if (kn < CH) {
      a0n = load_frag_a(A0 + kn, CH, lane);
      a1n = load_frag_a(A1 + kn, CH, lane);
      b0n = load_frag_b(Bt + (col0 +  0) * CH + kn, CH, lane);
      b1n = load_frag_b(Bt + (col0 + 16) * CH + kn, CH, lane);
      b2n = load_frag_b(Bt + (col0 + 32) * CH + kn, CH, lane);
      b3n = load_frag_b(Bt + (col0 + 48) * CH + kn, CH, lane);
    }
    acc[0][0] = wmma_bf16(a0, b0, acc[0][0]);
    acc[1][0] = wmma_bf16(a1, b0, acc[1][0]);
    acc[0][1] = wmma_bf16(a0, b1, acc[0][1]);
    acc[1][1] = wmma_bf16(a1, b1, acc[1][1]);
    acc[0][2] = wmma_bf16(a0, b2, acc[0][2]);
    acc[1][2] = wmma_bf16(a1, b2, acc[1][2]);
    acc[0][3] = wmma_bf16(a0, b3, acc[0][3]);
    acc[1][3] = wmma_bf16(a1, b3, acc[1][3]);
    a0 = a0n; a1 = a1n; b0 = b0n; b1 = b1n; b2 = b2n; b3 = b3n;
  }

  int half = lane >> 4, nloc = lane & 15;
#pragma unroll
  for (int j = 0; j < 4; ++j) {
    int n = col0 + j * 16 + nloc;
    float bv = bias[n];
#pragma unroll
    for (int mi = 0; mi < 2; ++mi) {
#pragma unroll
      for (int i = 0; i < 8; ++i) {
        int r = row0 + mi * 16 + i + 8 * half;
        if (r < ROWS) out[r * CH + n] = acc[mi][j][i] + bv;
      }
    }
  }
}

// ---------------------------------------------------------------- launch
extern "C" void kernel_launch(void* const* d_in, const int* in_sizes, int n_in,
                              void* d_out, int out_size, void* d_ws, size_t ws_size,
                              hipStream_t stream) {
  (void)in_sizes; (void)n_in; (void)out_size; (void)ws_size;

  const float*         x      = (const float*)d_in[0];
  const float*         cls    = (const float*)d_in[1];
  const unsigned char* maskp  = (const unsigned char*)d_in[2];
  // d_in[3], d_in[4]: mask_h/mask_w (constants 7, hardcoded)
  const float*         Wqkv   = (const float*)d_in[5];
  const float*         bqkv   = (const float*)d_in[6];
  const float*         Wo     = (const float*)d_in[7];
  const float*         bo     = (const float*)d_in[8];
  const float*         lk     = (const float*)d_in[9];
  const float*         lb     = (const float*)d_in[10];

  char* ws = (char*)d_ws;
  size_t off = 0;
  auto take = [&](size_t bytes) {
    char* p = ws + off;
    off = (off + bytes + 255) & ~(size_t)255;
    return p;
  };
  __bf16* Abf   = (__bf16*)take((size_t)ROWSPAD * CH * 2);
  __bf16* WqkvT = (__bf16*)take((size_t)768 * 256 * 2);
  __bf16* WoT   = (__bf16*)take((size_t)256 * 256 * 2);
  __bf16* Qb    = (__bf16*)take((size_t)BATCH * NH * LPAD * HD * 2);
  __bf16* Kb    = (__bf16*)take((size_t)BATCH * NH * LPAD * HD * 2);
  __bf16* Vt    = (__bf16*)take((size_t)BATCH * NH * HD * LPAD * 2);
  float*  Vsp   = (float*) take((size_t)BATCH * LWIN * CH * 4);
  float*  lepeB = (float*) take((size_t)BATCH * LWIN * CH * 4);
  float*  AO    = (float*) take((size_t)BATCH * NH * LPAD * HD * 4);
  __bf16* A2    = (__bf16*)take((size_t)ROWSPAD * CH * 2);

  prep_a_kernel<<<ROWSPAD, 256, 0, stream>>>(x, cls, Abf);
  prep_w_kernel<<<(768 * 256 + 256 * 256) / 256, 256, 0, stream>>>(Wqkv, Wo, WqkvT, WoT);
  gemm_qkv_kernel<<<dim3(ROWSPAD / 32, 3), 128, 0, stream>>>(Abf, WqkvT, bqkv,
                                                             Qb, Kb, Vt, Vsp);
  lepe_kernel<<<BATCH * LWIN, 256, 0, stream>>>(Vsp, lk, lb, lepeB);
  attn_kernel<<<(BATCH * NH * QTILES + 7) / 8, 256, 0, stream>>>(Qb, Kb, Vt, maskp, AO);
  build_a2_kernel<<<ROWSPAD, 256, 0, stream>>>(AO, lepeB, A2);
  gemm_out_kernel<<<dim3(ROWSPAD / 32, 1), 128, 0, stream>>>(A2, WoT, bo, (float*)d_out);
}
